// TripletGCNLayer_2963527434557
// MI455X (gfx1250) — matrix operand order repaired and checked by reading, not compiled, not run on hardware
//
#include <hip/hip_runtime.h>
#include <math.h>

#define B_ 16
#define N_ 1024
#define E_ 16384
#define D_ 256

typedef __attribute__((ext_vector_type(16))) __bf16 v16bf;
typedef __attribute__((ext_vector_type(8)))  float  v8f;

union BF16x16 { v16bf v; uint4 q[2]; };

__device__ __forceinline__ unsigned short f2bf(float f) {
  unsigned u = __float_as_uint(f);
  unsigned r = u + 0x7FFFu + ((u >> 16) & 1u);   // round-to-nearest-even
  return (unsigned short)(r >> 16);
}

__device__ __forceinline__ float silu_f(float x) {
  return x / (1.0f + __expf(-x));
}

// ---------------------------------------------------------------------------
// Pack fp32 weight (fi x fo, row-major) into bf16 WMMA B-tiles.
// Tile (kt,nt) covers K=[kt*32,+32), N=[nt*16,+16); tile order nt-major.
// Within a tile lane l owns 16 contiguous bf16: n = nt*16+(l&15),
// k = kt*32+(l>>4)*16+e  (contiguous K order, same as A: VGPR j = K 2j,2j+1; lanes
// 16..31 = K+16), so a lane fetches its 8 B-VGPRs with two b128 loads.
// ---------------------------------------------------------------------------
__global__ __launch_bounds__(256) void pack_w(const float* __restrict__ W,
                                              unsigned short* __restrict__ P,
                                              int fi, int fo) {
  int idx = blockIdx.x * 256 + threadIdx.x;
  if (idx >= fi * fo) return;
  int nkt    = fi >> 5;
  int tile   = idx >> 9;
  int within = idx & 511;
  int l = within >> 4;
  int e = within & 15;
  int nt = tile / nkt;
  int kt = tile - nt * nkt;
  int n = nt * 16 + (l & 15);
  int k = kt * 32 + (l >> 4) * 16 + e;
  P[idx] = f2bf(W[(size_t)k * fo + n]);
}

__global__ __launch_bounds__(256) void zero_f32(float* __restrict__ p, int n) {
  int i = blockIdx.x * 256 + threadIdx.x;
  if (i < n) p[i] = 0.0f;
}

// ---------------------------------------------------------------------------
// 2x2 register-blocked WMMA tile: two M-tiles (rows [0,16) and [16,32) of A)
// x two adjacent N-tiles (nt0, nt0+1). Each A/B fragment loaded once feeds
// two WMMAs -> halves both LDS and L2 operand traffic vs 1x1 tiling.
// ---------------------------------------------------------------------------
template <int NKT>
__device__ __forceinline__ void wmma_tile22(
    const unsigned short* __restrict__ A, int astr,
    const unsigned short* __restrict__ Wp, int nt0, int lane,
    v8f* a00, v8f* a01, v8f* a10, v8f* a11) {
  const int r    = lane & 15;
  const int koff = (lane >> 4) * 16;                // contiguous K: k = 16h + s
  const unsigned short* ar0 = A + r * astr + koff;
  const unsigned short* ar1 = ar0 + 16 * astr;
  const unsigned short* wl0 = Wp + (size_t)nt0 * NKT * 512 + lane * 16;
  const unsigned short* wl1 = wl0 + (size_t)NKT * 512;
#pragma unroll
  for (int kt = 0; kt < NKT; ++kt) {
    BF16x16 x0, x1, b0, b1;
    x0.q[0] = *(const uint4*)(ar0 + kt * 32);
    x0.q[1] = *(const uint4*)(ar0 + kt * 32 + 8);
    x1.q[0] = *(const uint4*)(ar1 + kt * 32);
    x1.q[1] = *(const uint4*)(ar1 + kt * 32 + 8);
    b0.q[0] = *(const uint4*)(wl0 + (size_t)kt * 512);
    b0.q[1] = *(const uint4*)(wl0 + (size_t)kt * 512 + 8);
    b1.q[0] = *(const uint4*)(wl1 + (size_t)kt * 512);
    b1.q[1] = *(const uint4*)(wl1 + (size_t)kt * 512 + 8);
    *a00 = __builtin_amdgcn_wmma_f32_16x16x32_bf16(false, x0.v, false, b0.v,
                                                   (short)0, *a00, false, false);
    *a01 = __builtin_amdgcn_wmma_f32_16x16x32_bf16(false, x0.v, false, b1.v,
                                                   (short)0, *a01, false, false);
    *a10 = __builtin_amdgcn_wmma_f32_16x16x32_bf16(false, x1.v, false, b0.v,
                                                   (short)0, *a10, false, false);
    *a11 = __builtin_amdgcn_wmma_f32_16x16x32_bf16(false, x1.v, false, b1.v,
                                                   (short)0, *a11, false, false);
  }
}

// 1x1 variant for the small node kernel.
template <int NKT>
__device__ __forceinline__ void wmma_gemm(const unsigned short* __restrict__ A,
                                          int astr,
                                          const unsigned short* __restrict__ Wp,
                                          int nt, int lane, v8f* acc) {
  const int r    = lane & 15;
  const int koff = (lane >> 4) * 16;
  const unsigned short* arow = A + r * astr + koff;
  const unsigned short* wl   = Wp + (size_t)nt * NKT * 512 + lane * 16;
#pragma unroll
  for (int kt = 0; kt < NKT; ++kt) {
    BF16x16 a, b;
    a.q[0] = *(const uint4*)(arow + kt * 32);
    a.q[1] = *(const uint4*)(arow + kt * 32 + 8);
    b.q[0] = *(const uint4*)(wl + (size_t)kt * 512);
    b.q[1] = *(const uint4*)(wl + (size_t)kt * 512 + 8);
    *acc = __builtin_amdgcn_wmma_f32_16x16x32_bf16(false, a.v, false, b.v,
                                                   (short)0, *acc, false, false);
  }
}

// ---------------------------------------------------------------------------
// Fused edge pipeline, 32 edges / block (two 16-row M-tiles), 8 waves.
// gather -> GEMM1+silu -> GEMM2+mask = msg -> atomic segment-sum ->
// GEMM3+silu -> GEMM4 + residual -> edge_out. msg never hits HBM.
// ---------------------------------------------------------------------------
#define ASTR 776   // 768 + 8 bf16 pad: 16B aligned, conflict-free b128 rows
#define HSTR 264   // 256 + 8 pad
#define MROWS 32

__global__ __launch_bounds__(256) void k_edge(
    const float* __restrict__ node_feat, const float* __restrict__ edge_feat,
    const int* __restrict__ edge_index, const unsigned char* __restrict__ edge_mask,
    const float* __restrict__ tb1, const float* __restrict__ tb2,
    const float* __restrict__ eb1, const float* __restrict__ eb2,
    const unsigned short* __restrict__ tw1p, const unsigned short* __restrict__ tw2p,
    const unsigned short* __restrict__ ew1p, const unsigned short* __restrict__ ew2p,
    float* __restrict__ agg, float* __restrict__ counts,
    float* __restrict__ edge_out) {
  __shared__ unsigned short A_tri[MROWS * ASTR];  // 32x768 bf16 triplet
  __shared__ unsigned short h_s[MROWS * HSTR];    // hidden / msg bf16 (reused)
  __shared__ unsigned short h2_s[MROWS * HSTR];   // edge-MLP hidden bf16
  __shared__ float msg_s[MROWS * HSTR];           // msg fp32 (for atomics)
  __shared__ int sidx[MROWS];
  __shared__ int didx[MROWS];
  __shared__ float maskf[MROWS];

  const int t = threadIdx.x;
  const int lane = t & 31;
  const int wave = t >> 5;
  const int blk = blockIdx.x;
  const int b  = blk >> 9;           // E_/32 = 512 tiles per batch
  const int e0 = (blk & 511) * 32;

  if (t < MROWS) {
    int e = e0 + t;
    sidx[t]  = edge_index[(size_t)b * 2 * E_ + e];
    didx[t]  = edge_index[(size_t)b * 2 * E_ + E_ + e];
    maskf[t] = edge_mask[(size_t)b * E_ + e] ? 1.0f : 0.0f;
  }
  __syncthreads();

  // ---- gather triplet rows (src | edge | dst), fp32 -> bf16 into LDS ----
  {
    const int m  = t >> 3;            // 8 threads per row, 32 rows
    const int c0 = (t & 7) * 96;
    const float* srow = node_feat + ((size_t)b * N_ + sidx[m]) * D_;
    const float* erow = edge_feat + ((size_t)b * E_ + e0 + m) * D_;
    const float* drow = node_feat + ((size_t)b * N_ + didx[m]) * D_;
#pragma unroll
    for (int j = 0; j < 24; ++j) {
      int c = c0 + j * 4;
      const float* p = (c < 256) ? (srow + c)
                     : (c < 512) ? (erow + (c - 256))
                                 : (drow + (c - 512));
      float4 v = *(const float4*)p;
      unsigned lo = (unsigned)f2bf(v.x) | ((unsigned)f2bf(v.y) << 16);
      unsigned hi = (unsigned)f2bf(v.z) | ((unsigned)f2bf(v.w) << 16);
      *(uint2*)&A_tri[m * ASTR + c] = make_uint2(lo, hi);
    }
  }
  __syncthreads();

  const int col   = lane & 15;
  const int mbase = 8 * (lane >> 4);
  const int nt0   = wave * 2;

  // ---- GEMM1: triplet (32x768) @ tw1 -> silu -> h ----
  {
    v8f a00 = {}, a01 = {}, a10 = {}, a11 = {};
    wmma_tile22<24>(A_tri, ASTR, tw1p, nt0, lane, &a00, &a01, &a10, &a11);
    const float bi0 = tb1[nt0 * 16 + col];
    const float bi1 = tb1[nt0 * 16 + 16 + col];
#pragma unroll
    for (int v = 0; v < 8; ++v) {
      h_s[(mbase + v) * HSTR + nt0 * 16 + col]        = f2bf(silu_f(a00[v] + bi0));
      h_s[(mbase + v) * HSTR + nt0 * 16 + 16 + col]   = f2bf(silu_f(a01[v] + bi1));
      h_s[(16 + mbase + v) * HSTR + nt0 * 16 + col]      = f2bf(silu_f(a10[v] + bi0));
      h_s[(16 + mbase + v) * HSTR + nt0 * 16 + 16 + col] = f2bf(silu_f(a11[v] + bi1));
    }
  }
  __syncthreads();

  // ---- GEMM2: h @ tw2 + tb2, * edge_mask -> msg (fp32 LDS) ----
  {
    v8f a00 = {}, a01 = {}, a10 = {}, a11 = {};
    wmma_tile22<8>(h_s, HSTR, tw2p, nt0, lane, &a00, &a01, &a10, &a11);
    const float bi0 = tb2[nt0 * 16 + col];
    const float bi1 = tb2[nt0 * 16 + 16 + col];
#pragma unroll
    for (int v = 0; v < 8; ++v) {
      int m0 = mbase + v, m1 = 16 + mbase + v;
      msg_s[m0 * HSTR + nt0 * 16 + col]      = (a00[v] + bi0) * maskf[m0];
      msg_s[m0 * HSTR + nt0 * 16 + 16 + col] = (a01[v] + bi1) * maskf[m0];
      msg_s[m1 * HSTR + nt0 * 16 + col]      = (a10[v] + bi0) * maskf[m1];
      msg_s[m1 * HSTR + nt0 * 16 + 16 + col] = (a11[v] + bi1) * maskf[m1];
    }
  }
  __syncthreads();

  // ---- segment-sum scatter (L2 atomics) + msg -> bf16 for edge MLP ----
  {
    const int m  = t >> 3;
    const int c0 = (t & 7) * 32;
    float* arow = agg + ((size_t)b * N_ + didx[m]) * D_;
#pragma unroll
    for (int j = 0; j < 32; ++j) {
      float v = msg_s[m * HSTR + c0 + j];
      h_s[m * HSTR + c0 + j] = f2bf(v);
      __hip_atomic_fetch_add(arow + c0 + j, v, __ATOMIC_RELAXED,
                             __HIP_MEMORY_SCOPE_AGENT);
    }
    if (t < MROWS)
      __hip_atomic_fetch_add(counts + (size_t)b * N_ + didx[t], maskf[t],
                             __ATOMIC_RELAXED, __HIP_MEMORY_SCOPE_AGENT);
  }
  __syncthreads();

  // ---- GEMM3: msg @ ew1 + eb1 -> silu -> h2 ----
  {
    v8f a00 = {}, a01 = {}, a10 = {}, a11 = {};
    wmma_tile22<8>(h_s, HSTR, ew1p, nt0, lane, &a00, &a01, &a10, &a11);
    const float bi0 = eb1[nt0 * 16 + col];
    const float bi1 = eb1[nt0 * 16 + 16 + col];
#pragma unroll
    for (int v = 0; v < 8; ++v) {
      h2_s[(mbase + v) * HSTR + nt0 * 16 + col]        = f2bf(silu_f(a00[v] + bi0));
      h2_s[(mbase + v) * HSTR + nt0 * 16 + 16 + col]   = f2bf(silu_f(a01[v] + bi1));
      h2_s[(16 + mbase + v) * HSTR + nt0 * 16 + col]      = f2bf(silu_f(a10[v] + bi0));
      h2_s[(16 + mbase + v) * HSTR + nt0 * 16 + 16 + col] = f2bf(silu_f(a11[v] + bi1));
    }
  }
  __syncthreads();

  // ---- GEMM4: h2 @ ew2 + eb2 + edge_feat -> edge_out ----
  {
    v8f a00 = {}, a01 = {}, a10 = {}, a11 = {};
    wmma_tile22<8>(h2_s, HSTR, ew2p, nt0, lane, &a00, &a01, &a10, &a11);
    const float bi0 = eb2[nt0 * 16 + col];
    const float bi1 = eb2[nt0 * 16 + 16 + col];
#pragma unroll
    for (int v = 0; v < 8; ++v) {
      int m0 = mbase + v, m1 = 16 + mbase + v;
      size_t o00 = ((size_t)b * E_ + e0 + m0) * D_ + nt0 * 16 + col;
      size_t o10 = ((size_t)b * E_ + e0 + m1) * D_ + nt0 * 16 + col;
      edge_out[o00]      = edge_feat[o00]      + a00[v] + bi0;
      edge_out[o00 + 16] = edge_feat[o00 + 16] + a01[v] + bi1;
      edge_out[o10]      = edge_feat[o10]      + a10[v] + bi0;
      edge_out[o10 + 16] = edge_feat[o10 + 16] + a11[v] + bi1;
    }
  }
}

// ---------------------------------------------------------------------------
// Node pipeline: agg/counts -> node MLP (WMMA) -> residual*mask -> LayerNorm.
// Block = 256 threads (8 waves) = 16 nodes (~2% of total FLOPs).
// ---------------------------------------------------------------------------
__global__ __launch_bounds__(256) void k_node(
    const float* __restrict__ node_feat, const unsigned char* __restrict__ node_mask,
    const float* __restrict__ nb1, const float* __restrict__ nb2,
    const float* __restrict__ gamma, const float* __restrict__ beta,
    const unsigned short* __restrict__ nw1p, const unsigned short* __restrict__ nw2p,
    const float* __restrict__ agg, const float* __restrict__ counts,
    float* __restrict__ node_out) {
  __shared__ unsigned short A_s[16 * HSTR];
  __shared__ unsigned short h_s[16 * HSTR];
  __shared__ float node_s[16 * HSTR];
  __shared__ float nmask[16];
  __shared__ float mu_s[16];
  __shared__ float rs_s[16];

  const int t = threadIdx.x;
  const int lane = t & 31;
  const int wave = t >> 5;
  const int b  = blockIdx.x >> 6;     // N_/16 = 64 tiles per batch
  const int n0 = (blockIdx.x & 63) * 16;

  if (t < 16) nmask[t] = node_mask[(size_t)b * N_ + n0 + t] ? 1.0f : 0.0f;

  {
    const int m  = t >> 4;
    const int c0 = (t & 15) * 16;
    const size_t row = (size_t)b * N_ + n0 + m;
    const float inv = 1.0f / fmaxf(counts[row], 1.0f);
    const float* arow = agg + row * D_;
#pragma unroll
    for (int j = 0; j < 16; ++j)
      A_s[m * HSTR + c0 + j] = f2bf(arow[c0 + j] * inv);
  }
  __syncthreads();

  const int col   = lane & 15;
  const int mbase = 8 * (lane >> 4);

#pragma unroll
  for (int i = 0; i < 2; ++i) {
    const int nt = wave * 2 + i;
    v8f acc = {};
    wmma_gemm<8>(A_s, HSTR, nw1p, nt, lane, &acc);
    const float bias = nb1[nt * 16 + col];
#pragma unroll
    for (int v = 0; v < 8; ++v)
      h_s[(mbase + v) * HSTR + nt * 16 + col] = f2bf(silu_f(acc[v] + bias));
  }
  __syncthreads();

#pragma unroll
  for (int i = 0; i < 2; ++i) {
    const int nt = wave * 2 + i;
    v8f acc = {};
    wmma_gemm<8>(h_s, HSTR, nw2p, nt, lane, &acc);
    const float bias = nb2[nt * 16 + col];
#pragma unroll
    for (int v = 0; v < 8; ++v) {
      int m = mbase + v;
      size_t off = ((size_t)b * N_ + n0 + m) * D_ + nt * 16 + col;
      node_s[m * HSTR + nt * 16 + col] =
          node_feat[off] + (acc[v] + bias) * nmask[m];
    }
  }
  __syncthreads();

  // ---- LayerNorm stats: wave w reduces rows 2w, 2w+1 via shuffles ----
#pragma unroll
  for (int rr = 0; rr < 2; ++rr) {
    const int r = wave * 2 + rr;
    float s = 0.0f, ss = 0.0f;
#pragma unroll
    for (int j = 0; j < 8; ++j) {
      float x = node_s[r * HSTR + lane + j * 32];
      s += x;
      ss += x * x;
    }
#pragma unroll
    for (int o = 16; o > 0; o >>= 1) {
      s  += __shfl_xor(s, o, 32);
      ss += __shfl_xor(ss, o, 32);
    }
    if (lane == 0) {
      float mu  = s * (1.0f / 256.0f);
      float var = ss * (1.0f / 256.0f) - mu * mu;
      mu_s[r] = mu;
      rs_s[r] = rsqrtf(var + 1e-5f);
    }
  }
  __syncthreads();

  {
    const int m  = t >> 4;
    const int c0 = (t & 15) * 16;
    const float mu = mu_s[m], rs = rs_s[m];
    float* orow = node_out + ((size_t)b * N_ + n0 + m) * D_;
#pragma unroll
    for (int j = 0; j < 16; ++j) {
      int c = c0 + j;
      orow[c] = (node_s[m * HSTR + c] - mu) * rs * gamma[c] + beta[c];
    }
  }
}

// ---------------------------------------------------------------------------
extern "C" void kernel_launch(void* const* d_in, const int* in_sizes, int n_in,
                              void* d_out, int out_size, void* d_ws,
                              size_t ws_size, hipStream_t stream) {
  (void)in_sizes; (void)n_in; (void)out_size; (void)ws_size;
  const float* node_feat = (const float*)d_in[0];
  const float* edge_feat = (const float*)d_in[1];
  const int*   edge_index = (const int*)d_in[2];
  const unsigned char* node_mask = (const unsigned char*)d_in[3];
  const unsigned char* edge_mask = (const unsigned char*)d_in[4];
  const float* tw1 = (const float*)d_in[5];
  const float* tb1 = (const float*)d_in[6];
  const float* tw2 = (const float*)d_in[7];
  const float* tb2 = (const float*)d_in[8];
  const float* nw1 = (const float*)d_in[9];
  const float* nb1 = (const float*)d_in[10];
  const float* nw2 = (const float*)d_in[11];
  const float* nb2 = (const float*)d_in[12];
  const float* ew1 = (const float*)d_in[13];
  const float* eb1 = (const float*)d_in[14];
  const float* ew2 = (const float*)d_in[15];
  const float* eb2 = (const float*)d_in[16];
  const float* gamma = (const float*)d_in[17];
  const float* beta  = (const float*)d_in[18];

  char* ws = (char*)d_ws;
  float* agg    = (float*)(ws);                             // 16 MiB
  float* counts = (float*)(ws + 16777216);                  // 64 KiB
  unsigned short* tw1p = (unsigned short*)(ws + 16842752);  // 384 KiB
  unsigned short* tw2p = (unsigned short*)(ws + 17235968);  // 128 KiB each
  unsigned short* ew1p = (unsigned short*)(ws + 17367040);
  unsigned short* ew2p = (unsigned short*)(ws + 17498112);
  unsigned short* nw1p = (unsigned short*)(ws + 17629184);
  unsigned short* nw2p = (unsigned short*)(ws + 17760256);

  float* node_out = (float*)d_out;
  float* edge_out = node_out + (size_t)B_ * N_ * D_;

  // agg and counts are contiguous -> single zero pass
  zero_f32<<<(B_ * N_ * D_ + B_ * N_ + 255) / 256, 256, 0, stream>>>(
      agg, B_ * N_ * D_ + B_ * N_);
  pack_w<<<(768 * 256) / 256, 256, 0, stream>>>(tw1, tw1p, 768, 256);
  pack_w<<<(256 * 256) / 256, 256, 0, stream>>>(tw2, tw2p, 256, 256);
  pack_w<<<(256 * 256) / 256, 256, 0, stream>>>(ew1, ew1p, 256, 256);
  pack_w<<<(256 * 256) / 256, 256, 0, stream>>>(ew2, ew2p, 256, 256);
  pack_w<<<(256 * 256) / 256, 256, 0, stream>>>(nw1, nw1p, 256, 256);
  pack_w<<<(256 * 256) / 256, 256, 0, stream>>>(nw2, nw2p, 256, 256);

  k_edge<<<B_ * (E_ / MROWS), 256, 0, stream>>>(
      node_feat, edge_feat, edge_index, edge_mask, tb1, tb2, eb1, eb2,
      tw1p, tw2p, ew1p, ew2p, agg, counts, edge_out);

  k_node<<<B_ * (N_ / 16), 256, 0, stream>>>(
      node_feat, node_mask, nb1, nb2, gamma, beta, nw1p, nw2p, agg, counts,
      node_out);
}
